// ProjectionLayer_33363305955536
// MI455X (gfx1250) — matrix operand order repaired and checked by reading, not compiled, not run on hardware
//
#include <hip/hip_runtime.h>
#include <hip/hip_bf16.h>
#include <stdint.h>

typedef __attribute__((ext_vector_type(16))) _Float16 v16h;
typedef __attribute__((ext_vector_type(8)))  float    v8f;

#define BATCH 32
#define MAXLEN 512
#define EMB 256
#define HID 1024
#define M_TOT (BATCH * MAXLEN)   // 16384

#define BPAD 8                    // f16 pad per LDS row (4 dwords) -> conflict-free b128 reads
#define BROW (EMB + BPAD)         // 264 f16 per LDS row

// ---------------------------------------------------------------------------
// SHA-256 / SHA-224 (single 64-byte block; message is <= 5 ASCII digits)
// ---------------------------------------------------------------------------
__device__ __constant__ uint32_t K256[64] = {
    0x428a2f98u,0x71374491u,0xb5c0fbcfu,0xe9b5dba5u,0x3956c25bu,0x59f111f1u,0x923f82a4u,0xab1c5ed5u,
    0xd807aa98u,0x12835b01u,0x243185beu,0x550c7dc3u,0x72be5d74u,0x80deb1feu,0x9bdc06a7u,0xc19bf174u,
    0xe49b69c1u,0xefbe4786u,0x0fc19dc6u,0x240ca1ccu,0x2de92c6fu,0x4a7484aau,0x5cb0a9dcu,0x76f988dau,
    0x983e5152u,0xa831c66du,0xb00327c8u,0xbf597fc7u,0xc6e00bf3u,0xd5a79147u,0x06ca6351u,0x14292967u,
    0x27b70a85u,0x2e1b2138u,0x4d2c6dfcu,0x53380d13u,0x650a7354u,0x766a0abbu,0x81c2c92eu,0x92722c85u,
    0xa2bfe8a1u,0xa81a664bu,0xc24b8b70u,0xc76c51a3u,0xd192e819u,0xd6990624u,0xf40e3585u,0x106aa070u,
    0x19a4c116u,0x1e376c08u,0x2748774cu,0x34b0bcb5u,0x391c0cb3u,0x4ed8aa4au,0x5b9cca4fu,0x682e6ff3u,
    0x748f82eeu,0x78a5636fu,0x84c87814u,0x8cc70208u,0x90befffau,0xa4506cebu,0xbef9a3f7u,0xc67178f2u
};

__device__ __constant__ uint32_t IV256[8] = {
    0x6a09e667u,0xbb67ae85u,0x3c6ef372u,0xa54ff53au,0x510e527fu,0x9b05688cu,0x1f83d9abu,0x5be0cd19u
};
__device__ __constant__ uint32_t IV224[8] = {
    0xc1059ed8u,0x367cd507u,0x3070dd17u,0xf70e5939u,0xffc00b31u,0x68581511u,0x64f98fa7u,0xbefa4fa4u
};

__device__ __forceinline__ uint32_t rotr32(uint32_t x, int n) {
    return (x >> n) | (x << (32 - n));
}

// Returns state word H0 after compressing one block with schedule words
// w0, w1, 0...0, w15.  digest byte 0 == (H0 >> 24) & 0xff.
__device__ uint32_t sha2_h0(uint32_t w0, uint32_t w1, uint32_t w15, const uint32_t* iv) {
    uint32_t w[64];
    w[0] = w0; w[1] = w1;
    for (int i = 2; i < 15; ++i) w[i] = 0u;
    w[15] = w15;
    for (int i = 16; i < 64; ++i) {
        uint32_t x = w[i - 15], y = w[i - 2];
        uint32_t s0 = rotr32(x, 7) ^ rotr32(x, 18) ^ (x >> 3);
        uint32_t s1 = rotr32(y, 17) ^ rotr32(y, 19) ^ (y >> 10);
        w[i] = w[i - 16] + s0 + w[i - 7] + s1;
    }
    uint32_t a = iv[0], b = iv[1], c = iv[2], d = iv[3];
    uint32_t e = iv[4], f = iv[5], g = iv[6], h = iv[7];
    for (int i = 0; i < 64; ++i) {
        uint32_t S1 = rotr32(e, 6) ^ rotr32(e, 11) ^ rotr32(e, 25);
        uint32_t ch = (e & f) ^ (~e & g);
        uint32_t t1 = h + S1 + ch + K256[i] + w[i];
        uint32_t S0 = rotr32(a, 2) ^ rotr32(a, 13) ^ rotr32(a, 22);
        uint32_t mj = (a & b) ^ (a & c) ^ (b & c);
        uint32_t t2 = S0 + mj;
        h = g; g = f; f = e; e = d + t1;
        d = c; c = b; b = a; a = t1 + t2;
    }
    return iv[0] + a;
}

// h1|h2<<8 per token -> ws hashes array
__global__ __launch_bounds__(256) void hash_kernel(const int* __restrict__ ids,
                                                   unsigned short* __restrict__ hashes,
                                                   int n) {
    int i = blockIdx.x * blockDim.x + threadIdx.x;
    if (i >= n) return;
    uint32_t v = (uint32_t)ids[i];

    // decimal digits, least-significant first
    char dig[8];
    int nd = 0;
    uint32_t t = v;
    do { dig[nd++] = (char)('0' + (t % 10u)); t /= 10u; } while (t != 0u);

    uint32_t b0 = 0u, b1 = 0u;   // big-endian schedule words 0 and 1
    #pragma unroll
    for (int j = 0; j < 8; ++j) {
        if (j < nd) {
            uint32_t c = (uint32_t)(unsigned char)dig[nd - 1 - j];
            if (j < 4) b0 |= c << (24 - 8 * j);
            else       b1 |= c << (24 - 8 * (j - 4));
        }
    }
    if (nd < 4) b0 |= 0x80u << (24 - 8 * nd);
    else        b1 |= 0x80u << (24 - 8 * (nd - 4));
    uint32_t w15 = (uint32_t)(nd * 8);   // bit length

    uint32_t h1 = (sha2_h0(b0, b1, w15, IV256) >> 24) & 0xffu;
    uint32_t h2 = (sha2_h0(b0, b1, w15, IV224) >> 24) & 0xffu;
    hashes[i] = (unsigned short)(h1 | (h2 << 8));
}

// W f32[HID][EMB] -> f16, done once; values feed WMMA B fragments
__global__ __launch_bounds__(256) void convert_w(const float* __restrict__ W,
                                                 _Float16* __restrict__ W16, int n) {
    int i = blockIdx.x * blockDim.x + threadIdx.x;
    if (i < n) W16[i] = (_Float16)W[i];
}

// ---------------------------------------------------------------------------
// GEMM: out[m,n] = sum_k ((h1[m] + k*h2[m]) & 255) * W[n,k] + b[n]
// Block: 256 threads = 8 waves, 128(M) x 128(N) tile; wave = 16(M) x 128(N).
// B panel (128 rows x 256 f16 of W) staged ONCE per block into padded LDS and
// shared by all 8 waves; A synthesized in-register (zero memory bytes);
// 8 x v_wmma_f32_16x16x32_f16 per K-step with f32 accumulation.
// ---------------------------------------------------------------------------
__global__ __launch_bounds__(256) void gemm_kernel(const _Float16* __restrict__ W16,
                                                   const unsigned short* __restrict__ hashes,
                                                   const float* __restrict__ bias,
                                                   float* __restrict__ out) {
    __shared__ _Float16 Bs[128 * BROW];   // 128*264*2 = 67584 B of the 320 KB WGP LDS

    const int tid  = threadIdx.x;
    const int lane = tid & 31;
    const int wave = tid >> 5;
    const int hi   = lane >> 4;      // which half of the wave
    const int lr   = lane & 15;
    const int m0   = blockIdx.x * 128 + wave * 16;
    const int n0   = blockIdx.y * 128;

    // --- cooperative stage of W panel: 128 rows x 256 f16 = 4096 x 16B chunks
    {
        const uint4* __restrict__ src = (const uint4*)(W16 + ((size_t)n0 << 8));
        #pragma unroll
        for (int c = 0; c < 16; ++c) {
            const int idx = c * 256 + tid;       // chunk index 0..4095
            const int row = idx >> 5;            // /32 chunks per row
            const int c8  = idx & 31;            // chunk within row (8 f16 each)
            uint4 d = src[row * 32 + c8];
            *(uint4*)(Bs + row * BROW + c8 * 8) = d;
        }
    }

    const unsigned short hv = hashes[m0 + lr];
    const int h1 = hv & 255;
    const int h2 = hv >> 8;

    __syncthreads();

    v8f acc[8];
    #pragma unroll
    for (int nt = 0; nt < 8; ++nt) acc[nt] = (v8f){0.f,0.f,0.f,0.f,0.f,0.f,0.f,0.f};

    #pragma unroll
    for (int kk = 0; kk < 8; ++kk) {           // K = 256 in steps of 32
        // Build A fragment in contiguous K order (matches B fragment):
        // lanes 0-15 : VGPR j -> K pair (2j,2j+1), K 0-15
        // lanes 16-31: same rows, K 16-31.
        v16h afrag;
        #pragma unroll
        for (int j = 0; j < 8; ++j) {
            const int kb = kk * 32 + hi * 16 + 2 * j;
            afrag[2 * j]     = (_Float16)((h1 + kb * h2) & 255);
            afrag[2 * j + 1] = (_Float16)((h1 + (kb + 1) * h2) & 255);
        }

        #pragma unroll
        for (int nt = 0; nt < 8; ++nt) {
            // B fragment from LDS: lane lr = column (W row), lane-half selects
            // K 0-15 / 16-31 of this 32-wide K step.  Row stride 264 f16 =
            // 132 dwords == 4 (mod 64) banks => each ds_load_b128 hits every
            // bank exactly twice (minimum possible for 512 B/instr).
            const _Float16* bp = Bs + (nt * 16 + lr) * BROW
                                    + kk * 32 + hi * 16;
            v16h bfrag;
            *(uint4*)&bfrag       = *(const uint4*)bp;        // ds_load_b128
            *((uint4*)&bfrag + 1) = *(const uint4*)(bp + 8);  // ds_load_b128
            acc[nt] = __builtin_amdgcn_wmma_f32_16x16x32_f16(
                false, afrag, false, bfrag, (short)0, acc[nt], false, false);
        }
    }

    // Epilogue: C layout — VGPR v holds (row m0+v | m0+v+8 for hi half), col = lane&15.
    #pragma unroll
    for (int nt = 0; nt < 8; ++nt) {
        const int col = n0 + nt * 16 + lr;
        const float bb = bias[col];
        #pragma unroll
        for (int v = 0; v < 8; ++v) {
            const int row = m0 + v + hi * 8;
            out[(size_t)row * HID + col] = acc[nt][v] + bb;
        }
    }
}

extern "C" void kernel_launch(void* const* d_in, const int* in_sizes, int n_in,
                              void* d_out, int out_size, void* d_ws, size_t ws_size,
                              hipStream_t stream) {
    const int*   ids = (const int*)d_in[0];     // [32,512]
    const float* W   = (const float*)d_in[1];   // [1024,256]
    const float* b   = (const float*)d_in[2];   // [1024]
    float*       out = (float*)d_out;           // [32,512,1024]

    // workspace layout: [0, 512KB) f16 W, [512KB, 512KB+32KB) token hashes
    _Float16*       W16    = (_Float16*)d_ws;
    unsigned short* hashes = (unsigned short*)((char*)d_ws + (size_t)HID * EMB * sizeof(_Float16));

    hash_kernel<<<M_TOT / 256, 256, 0, stream>>>(ids, hashes, M_TOT);
    convert_w<<<(HID * EMB) / 256, 256, 0, stream>>>(W, W16, HID * EMB);

    dim3 grid(M_TOT / 128, HID / 128, 1);       // 128 x 8 blocks
    gemm_kernel<<<grid, 256, 0, stream>>>(W16, hashes, b, out);

    (void)in_sizes; (void)n_in; (void)out_size; (void)ws_size;
}
